// PacCRF_601295421593
// MI455X (gfx1250) — matrix operand order repaired and checked by reading, not compiled, not run on hardware
//
#include <hip/hip_runtime.h>
#include <hip/hip_bf16.h>

typedef __attribute__((ext_vector_type(16))) _Float16 v16h;
typedef __attribute__((ext_vector_type(8)))  _Float16 v8h;
typedef __attribute__((ext_vector_type(8)))  float    v8f;

#define BB     2
#define CC     21
#define HH     512
#define WW     512
#define CF     3
#define NTAP   9
#define QH     (HH + 2)
#define QW     (WW + 2)
#define QC     32             // padded channel stride in Q buffer
#define NSTEPS 5

#define KERN_BYTES ((size_t)BB * NTAP * HH * WW * 2)     // f16 kern: 9,437,184
#define QBUF_BYTES ((size_t)BB * QH * QW * QC * 2)       // 33,817,088
#define BFRAG_ELTS (NTAP * 2 * 32 * 16)                  // 9216 halves

// ---------------------------------------------------------------------------
// 1) Edge-adaptive Gaussian kernel (constant across steps), stored f16:
//    kern[b,t,h,w] = exp(-0.5*||f_j - f_i||^2), zero-padded neighbors.
// ---------------------------------------------------------------------------
__global__ void k_edge(const float* __restrict__ f, _Float16* __restrict__ kern) {
    int idx = blockIdx.x * blockDim.x + threadIdx.x;
    if (idx >= BB * HH * WW) return;
    int w = idx % WW, h = (idx / WW) % HH, b = idx / (WW * HH);
    const float* fb = f + (size_t)b * CF * HH * WW;
    size_t hw = (size_t)h * WW + w;
    float c0 = fb[0 * HH * WW + hw];
    float c1 = fb[1 * HH * WW + hw];
    float c2 = fb[2 * HH * WW + hw];
#pragma unroll
    for (int t = 0; t < NTAP; ++t) {
        int hy = h + t / 3 - 1, wx = w + t % 3 - 1;
        float n0 = 0.f, n1 = 0.f, n2 = 0.f;
        if (hy >= 0 && hy < HH && wx >= 0 && wx < WW) {
            size_t nhw = (size_t)hy * WW + wx;
            n0 = fb[0 * HH * WW + nhw];
            n1 = fb[1 * HH * WW + nhw];
            n2 = fb[2 * HH * WW + nhw];
        }
        float d0 = n0 - c0, d1 = n1 - c1, d2 = n2 - c2;
        kern[((size_t)(b * NTAP + t) * HH + h) * WW + w] =
            (_Float16)__expf(-0.5f * (d0 * d0 + d1 * d1 + d2 * d2));
    }
}

// ---------------------------------------------------------------------------
// 2) Zero the Q buffer (halo rows/cols and pad channels 21..31 stay zero).
// ---------------------------------------------------------------------------
__global__ void k_zero16(uint4* __restrict__ p, unsigned n16) {
    unsigned i = blockIdx.x * blockDim.x + threadIdx.x;
    if (i < n16) p[i] = make_uint4(0u, 0u, 0u, 0u);
}

// ---------------------------------------------------------------------------
// 3) Per-tap WMMA B-fragments (f16): B_t[k = i, n = o] = weight[o][i][t],
//    K padded 21->32 with zeros. Per-lane layout (ISA 7.12.2): lane holds
//    col n = lane&15; half p -> K = (lane>>4)*8 + p (p<8), 16 + (lane>>4)*8 + (p-8) (p>=8).
//    Fragment f = ((t*2 + ntile)*32 + lane)*16 + p.
// ---------------------------------------------------------------------------
__global__ void k_bfrag(const float* __restrict__ wgt, _Float16* __restrict__ bf) {
    int f = blockIdx.x * blockDim.x + threadIdx.x;
    if (f >= BFRAG_ELTS) return;
    int p    = f & 15;
    int lane = (f >> 4) & 31;
    int nt   = (f >> 9) & 1;
    int t    = f / (16 * 32 * 2);
    int n    = nt * 16 + (lane & 15);
    int k    = (p < 8) ? (lane >> 4) * 8 + p : 16 + (lane >> 4) * 8 + (p - 8);
    float v = 0.f;
    if (k < CC && n < CC)
        v = wgt[(n * CC + k) * NTAP + t];     // weight[o][i][ty][tx] flat
    bf[f] = (_Float16)v;
}

// ---------------------------------------------------------------------------
// 4) Channel softmax of logQ -> Q (f16, channel-last, halo layout).
// ---------------------------------------------------------------------------
__global__ void k_softmax(const float* __restrict__ src, _Float16* __restrict__ qbuf) {
    int idx = blockIdx.x * blockDim.x + threadIdx.x;
    if (idx >= BB * HH * WW) return;
    int w = idx % WW, h = (idx / WW) % HH, b = idx / (WW * HH);
    const float* s = src + (size_t)b * CC * HH * WW + (size_t)h * WW + w;
    float v[CC];
    float mx = -3.0e38f;
#pragma unroll
    for (int c = 0; c < CC; ++c) { v[c] = s[(size_t)c * HH * WW]; mx = fmaxf(mx, v[c]); }
    float sum = 0.f;
#pragma unroll
    for (int c = 0; c < CC; ++c) { v[c] = __expf(v[c] - mx); sum += v[c]; }
    float inv = 1.0f / sum;
    _Float16* q = qbuf + (((size_t)b * QH + (h + 1)) * QW + (w + 1)) * QC;
#pragma unroll
    for (int c = 0; c < CC; ++c) q[c] = (_Float16)(v[c] * inv);
}

// ---------------------------------------------------------------------------
// 5) Message passing via per-tap WMMA chain + logQ update.
//    1 wave = 16-pixel strip (row h, cols w0..w0+15). For each of 9 taps:
//      - A fragment = Q channel vectors of the shifted pixels, loaded with
//        two 16B global loads per lane (channel-last QC=32 layout matches
//        the ISA A-fragment k positions), scaled by the per-row scalar
//        kern[t,h,w0+m] with packed f16 multiplies,
//      - acc = wmma_f32_16x16x32_f16(a, B_t, acc) for 2 N-tiles.
//    18 WMMAs per strip; no LDS, no gather loop.
//    logQ = unary * uw - msg
// ---------------------------------------------------------------------------
__global__ __launch_bounds__(256)
void k_msg(const float* __restrict__ unary, const float* __restrict__ uw_p,
           const _Float16* __restrict__ kern, const _Float16* __restrict__ qbuf,
           const _Float16* __restrict__ bfrag, float* __restrict__ out) {
    const int lane = threadIdx.x & 31;
    const int wv   = threadIdx.x >> 5;
    const int tile = blockIdx.x * 8 + wv;     // [0, B*H*W/16)
    const int wt = tile % (WW / 16);
    const int h  = (tile / (WW / 16)) % HH;
    const int b  = tile / ((WW / 16) * HH);
    const int w0 = wt * 16;

    const int g    = lane >> 4;
    const int mrow = lane & 15;

    const _Float16* qb = qbuf + (size_t)b * QH * QW * QC;
    const _Float16* kb = kern + (size_t)b * NTAP * HH * WW + (size_t)h * WW + w0 + mrow;

    v8f acc0 = {}, acc1 = {};
#pragma unroll
    for (int t = 0; t < NTAP; ++t) {
        // A fragment: row m = mrow at shifted pixel; halves k = g*8+p / 16+g*8+p
        const int hy  = h + t / 3;                 // halo-indexed (dy+1)
        const int wx  = w0 + mrow + t % 3;         // halo-indexed (dx+1)
        const _Float16* ap = qb + ((size_t)hy * QW + wx) * QC + g * 8;
        v8h lo = *(const v8h*)ap;
        v8h hi = *(const v8h*)(ap + 16);

        // per-row pixel-adaptive scale
        _Float16 kv = kb[(size_t)t * HH * WW];
        v8h kvv = {kv, kv, kv, kv, kv, kv, kv, kv};
        lo = lo * kvv;
        hi = hi * kvv;
        v16h a;
#pragma unroll
        for (int p = 0; p < 8; ++p) { a[p] = lo[p]; a[p + 8] = hi[p]; }

        v16h b0 = *(const v16h*)(bfrag + ((size_t)(t * 2 + 0) * 32 + lane) * 16);
        v16h b1 = *(const v16h*)(bfrag + ((size_t)(t * 2 + 1) * 32 + lane) * 16);
        acc0 = __builtin_amdgcn_wmma_f32_16x16x32_f16(false, a, false, b0,
                                                      (short)0, acc0, false, false);
        acc1 = __builtin_amdgcn_wmma_f32_16x16x32_f16(false, a, false, b1,
                                                      (short)0, acc1, false, false);
    }

    // ---- writeback: lane -> (o = ntile*16 + (lane&15), pixels w0 + g*8 .. +7)
    const float uw = uw_p[0];
    const int m0 = g * 8;
    {
        int o = mrow;                                     // 0..15, all valid
        size_t base = ((size_t)(b * CC + o) * HH + h) * WW + w0 + m0;
        const float4* up = (const float4*)(unary + base);
        float4* op = (float4*)(out + base);
        float4 u0 = up[0], u1 = up[1], r0, r1;
        r0.x = uw * u0.x - acc0[0]; r0.y = uw * u0.y - acc0[1];
        r0.z = uw * u0.z - acc0[2]; r0.w = uw * u0.w - acc0[3];
        r1.x = uw * u1.x - acc0[4]; r1.y = uw * u1.y - acc0[5];
        r1.z = uw * u1.z - acc0[6]; r1.w = uw * u1.w - acc0[7];
        op[0] = r0; op[1] = r1;
    }
    if (mrow < CC - 16) {                                 // o = 16..20
        int o = 16 + mrow;
        size_t base = ((size_t)(b * CC + o) * HH + h) * WW + w0 + m0;
        const float4* up = (const float4*)(unary + base);
        float4* op = (float4*)(out + base);
        float4 u0 = up[0], u1 = up[1], r0, r1;
        r0.x = uw * u0.x - acc1[0]; r0.y = uw * u0.y - acc1[1];
        r0.z = uw * u0.z - acc1[2]; r0.w = uw * u0.w - acc1[3];
        r1.x = uw * u1.x - acc1[4]; r1.y = uw * u1.y - acc1[5];
        r1.z = uw * u1.z - acc1[6]; r1.w = uw * u1.w - acc1[7];
        op[0] = r0; op[1] = r1;
    }
}

// ---------------------------------------------------------------------------
extern "C" void kernel_launch(void* const* d_in, const int* in_sizes, int n_in,
                              void* d_out, int out_size, void* d_ws, size_t ws_size,
                              hipStream_t stream) {
    const float* unary = (const float*)d_in[0];   // [2,21,512,512]
    const float* edge  = (const float*)d_in[1];   // [2,3,512,512]
    const float* wgt   = (const float*)d_in[2];   // [21,21,3,3]
    const float* uw    = (const float*)d_in[3];   // scalar
    float* out = (float*)d_out;                   // logQ buffer + final output

    char* ws = (char*)d_ws;
    _Float16* kern  = (_Float16*)ws;
    _Float16* qbuf  = (_Float16*)(ws + KERN_BYTES);
    _Float16* bfrag = (_Float16*)(ws + KERN_BYTES + QBUF_BYTES);

    const int npix = BB * HH * WW;

    // one-time per launch: edge kernel, Q buffer zero, B fragments
    k_edge<<<(npix + 255) / 256, 256, 0, stream>>>(edge, kern);
    unsigned q16 = (unsigned)(QBUF_BYTES / 16);
    k_zero16<<<(q16 + 255) / 256, 256, 0, stream>>>((uint4*)qbuf, q16);
    k_bfrag<<<(BFRAG_ELTS + 255) / 256, 256, 0, stream>>>(wgt, bfrag);

    const int ntiles  = BB * HH * (WW / 16);      // 32768 waves
    const int nblocks = ntiles / 8;               // 8 waves per block

    for (int s = 0; s < NSTEPS; ++s) {
        const float* src = (s == 0) ? unary : out;
        k_softmax<<<(npix + 255) / 256, 256, 0, stream>>>(src, qbuf);
        k_msg<<<nblocks, 256, 0, stream>>>(unary, uw, kern, qbuf, bfrag, out);
    }
}
